// Conv_DCFD_8469675507911
// MI455X (gfx1250) — compile-verified, matrix operand order checked by reading, not run on hardware
//
#include <hip/hip_runtime.h>
#include <hip/hip_bf16.h>
#include <math.h>

// MI455X (gfx1250) DCFD forward — round 3.
// Compute-bound (~15.5 GFLOP vs ~2us HBM time at 23.3 TB/s); all GEMM stages on
// v_wmma_f32_16x16x32_f16 (wave32), K reordered as (tap, c) so A-fragments are
// 2x ds_load_b128 from an LDS halo and B-fragments are 2x global_load_b128 from
// pre-transposed f16 weights. New this round: float4-vectorized conv1 halo
// staging, async global->LDS (ASYNCcnt) staging of the conv2 halo, and
// global_prefetch on the coef stream.

#define USE_ASYNC_LDS 1   // inline-asm global_load_async_to_lds_b128 path

typedef __attribute__((ext_vector_type(16))) _Float16 v16h;
typedef __attribute__((ext_vector_type(8)))  _Float16 v8h;
typedef __attribute__((ext_vector_type(8)))  float    v8f;

__device__ __forceinline__ v16h cat16(v8h lo, v8h hi) {
  return __builtin_shufflevector(lo, hi, 0,1,2,3,4,5,6,7,8,9,10,11,12,13,14,15);
}

// ---------------------------------------------------------------- f32 -> f16
__global__ __launch_bounds__(256) void k_cvt(const float* __restrict__ a,
                                             _Float16* __restrict__ b, int n) {
  int i = blockIdx.x * 256 + threadIdx.x;
  if (i < n) b[i] = (_Float16)a[i];
}

// f32 -> f16 with (c,tap) -> (tap,c) transpose of 3x3 conv weights.
__global__ __launch_bounds__(256) void k_cvt_w(const float* __restrict__ a,
                                               _Float16* __restrict__ b,
                                               int OC, int CI) {
  int i = blockIdx.x * 256 + threadIdx.x;
  if (i >= OC * CI * 9) return;
  int oc = i / (CI * 9); int r = i - oc * CI * 9;     // r = c*9 + tap
  int c = r / 9; int tap = r - c * 9;
  b[oc * CI * 9 + tap * CI + c] = (_Float16)a[i];
}

// ----------------------------------------------- conv1 (3x3, 128->64) + BN + tanh
// One block per (n,h) row: M=64 pixels (4 tiles) x N=64 (4 tiles), K=1152.
// K order: k = tap*128 + c.  LDS halo: [3 rows][66 cols][128 c] padded to 136.
#define X1STR 136
__global__ __launch_bounds__(256) void k_conv1(
    const float* __restrict__ x, const _Float16* __restrict__ w1t,
    const float* __restrict__ cb, const float* __restrict__ g,
    const float* __restrict__ be, const float* __restrict__ mu,
    const float* __restrict__ vr, _Float16* __restrict__ h1) {
  __shared__ _Float16 xl[198 * X1STR];                  // 53.9 KB
  const int blk = blockIdx.x;                           // 512 = n*64 + h
  const int n = blk >> 6, hh = blk & 63;
  const float* xb = x + n * (128 * 4096);
  // zero the two border columns (col 0 and 65) for all rows/channels
  for (int i = threadIdx.x; i < 3 * 2 * 128; i += 256) {
    int c = i & 127; int q = i >> 7;                    // q = row*2 + side
    int row = q >> 1, side = (q & 1) ? 65 : 0;
    xl[(row * 66 + side) * X1STR + c] = (_Float16)0.0f;
  }
  // interior: float4 vector loads over 4-column groups (coalesced, 16B aligned)
  for (int row = 0; row < 3; ++row) {
    int yy = hh + row - 1;
    if ((unsigned)yy < 64u) {
      const float* xr = xb + yy * 64;
      for (int i = threadIdx.x; i < 128 * 16; i += 256) {
        int c = i >> 4, gq = i & 15;
        float4 v = *(const float4*)(xr + c * 4096 + gq * 4);
        int base = (row * 66 + 1 + gq * 4) * X1STR + c;
        xl[base]             = (_Float16)v.x;
        xl[base + X1STR]     = (_Float16)v.y;
        xl[base + 2 * X1STR] = (_Float16)v.z;
        xl[base + 3 * X1STR] = (_Float16)v.w;
      }
    } else {
      for (int i = threadIdx.x; i < 64 * 128; i += 256) {
        int c = i & 127, col = (i >> 7) + 1;
        xl[(row * 66 + col) * X1STR + c] = (_Float16)0.0f;
      }
    }
  }
  __syncthreads();
  const int wv = threadIdx.x >> 5, lane = threadIdx.x & 31;
  const int mrow = lane & 15, hi = lane >> 4;
  const int mtile = wv & 3, ng = wv >> 2;               // ng: ntiles {0,1} / {2,3}
  const int p = mtile * 16 + mrow;                      // pixel (w) for A fragment
  v8f acc[2]; acc[0] = v8f{}; acc[1] = v8f{};
  for (int tap = 0; tap < 9; ++tap) {
    const int dy = tap / 3, dx = tap - dy * 3;
    const _Float16* arow = &xl[(dy * 66 + p + dx) * X1STR];
    const _Float16* wrow = w1t + tap * 128;
    for (int cs = 0; cs < 4; ++cs) {                    // 4 chunks of 32 channels
      const int cbk = cs * 32 + 8 * hi;
      const _Float16* ap = arow + cbk;
      v16h A = cat16(*(const v8h*)ap, *(const v8h*)(ap + 16));
#pragma unroll
      for (int t = 0; t < 2; ++t) {
        const _Float16* bp = wrow + ((ng * 2 + t) * 16 + mrow) * 1152 + cbk;
        v16h B = cat16(*(const v8h*)bp, *(const v8h*)(bp + 16));
        acc[t] = __builtin_amdgcn_wmma_f32_16x16x32_f16(false, A, false, B,
                                                        (short)0, acc[t], false, false);
      }
    }
  }
  const int rowbase = (n * 64 + hh) * 64;
#pragma unroll
  for (int t = 0; t < 2; ++t) {
    int ch = (ng * 2 + t) * 16 + mrow;
    float s = g[ch] * rsqrtf(vr[ch] + 1e-5f);
    float t0 = (cb[ch] - mu[ch]) * s + be[ch];
#pragma unroll
    for (int r = 0; r < 8; ++r) {
      int w = mtile * 16 + r + 8 * hi;
      h1[(rowbase + w) * 64 + ch] = (_Float16)tanhf(acc[t][r] * s + t0);
    }
  }
}

// ----------------------------------------------- conv2 (3x3, 64->72) + BN + tanh
// K = tap*64 + c (576). LDS halo: [3][66][64 c] padded to 72. N tiles: 5.
// Halo staged with async global->LDS b128 copies (ASYNCcnt) where available.
#define X2STR 72
__global__ __launch_bounds__(256) void k_conv2(
    const _Float16* __restrict__ h1, const _Float16* __restrict__ w2t,
    const float* __restrict__ cb, const float* __restrict__ g,
    const float* __restrict__ be, const float* __restrict__ mu,
    const float* __restrict__ vr, _Float16* __restrict__ h2) {
  __shared__ _Float16 xl[198 * X2STR];                  // 28.5 KB
  const int blk = blockIdx.x;
  const int n = blk >> 6, hh = blk & 63;
  for (int i = threadIdx.x; i < 198 * 8; i += 256) {    // 16B-granular halo copy
    int cg = i & 7, rc = i >> 3;
    int row = (rc >= 132) ? 2 : (rc >= 66 ? 1 : 0);
    int col = rc - row * 66;
    int yy = hh + row - 1, xx = col - 1;
    _Float16* dst = &xl[rc * X2STR + cg * 8];
    if ((unsigned)yy < 64u && (unsigned)xx < 64u) {
      const _Float16* src = h1 + ((n * 64 + yy) * 64 + xx) * 64 + cg * 8;
#if USE_ASYNC_LDS
      unsigned laddr = (unsigned)(size_t)(void*)dst;    // generic LDS addr[31:0] == LDS offset
      asm volatile("global_load_async_to_lds_b128 %0, %1, off"
                   :: "v"(laddr), "v"((unsigned long long)(size_t)src)
                   : "memory");
#else
      *(v8h*)dst = *(const v8h*)src;
#endif
    } else {
      v8h z = {};
      *(v8h*)dst = z;
    }
  }
#if USE_ASYNC_LDS
  asm volatile("s_wait_asynccnt 0x0" ::: "memory");
#endif
  __syncthreads();
  const int wv = threadIdx.x >> 5, lane = threadIdx.x & 31;
  const int mrow = lane & 15, hi = lane >> 4;
  const int mtile = wv & 3, ng = wv >> 2;               // ng: ntiles {0,1} / {2,3,4}
  const int T = ng ? 3 : 2, nbase = ng ? 2 : 0;
  const int p = mtile * 16 + mrow;
  v8f acc[3]; acc[0] = v8f{}; acc[1] = v8f{}; acc[2] = v8f{};
  for (int tap = 0; tap < 9; ++tap) {
    const int dy = tap / 3, dx = tap - dy * 3;
    const _Float16* arow = &xl[(dy * 66 + p + dx) * X2STR];
    const _Float16* wrow = w2t + tap * 64;
    for (int cs = 0; cs < 2; ++cs) {
      const int cbk = cs * 32 + 8 * hi;
      const _Float16* ap = arow + cbk;
      v16h A = cat16(*(const v8h*)ap, *(const v8h*)(ap + 16));
#pragma unroll
      for (int t = 0; t < 3; ++t) {
        if (t < T) {                                    // wave-uniform branch
          int oc = (nbase + t) * 16 + mrow; if (oc > 71) oc = 71;
          const _Float16* bp = wrow + oc * 576 + cbk;
          v16h B = cat16(*(const v8h*)bp, *(const v8h*)(bp + 16));
          acc[t] = __builtin_amdgcn_wmma_f32_16x16x32_f16(false, A, false, B,
                                                          (short)0, acc[t], false, false);
        }
      }
    }
  }
  const int rowbase = (n * 64 + hh) * 64;
#pragma unroll
  for (int t = 0; t < 3; ++t) {
    if (t < T) {
      int ch = (nbase + t) * 16 + mrow;
      int chc = ch < 72 ? ch : 71;
      float s = g[chc] * rsqrtf(vr[chc] + 1e-5f);
      float t0 = (cb[chc] - mu[chc]) * s + be[chc];
#pragma unroll
      for (int r = 0; r < 8; ++r) {
        int w = mtile * 16 + r + 8 * hi;
        if (ch < 72) h2[(rowbase + w) * 72 + ch] = (_Float16)tanhf(acc[t][r] * s + t0);
      }
    }
  }
}

// ------------------------------------------------- basis projection (12 -> 25)
__global__ __launch_bounds__(256) void k_proj(const _Float16* __restrict__ h2,
                                              const float* __restrict__ fb,
                                              _Float16* __restrict__ bp) {
  int idx = blockIdx.x * 256 + threadIdx.x;             // 32768 * 150
  if (idx >= 32768 * 150) return;
  int p = idx / 150, v = idx - p * 150;
  int m = v / 25, l = v - m * 25;
  const _Float16* hp = h2 + p * 72 + m * 12;
  float s = 0.0f;
#pragma unroll
  for (int t = 0; t < 12; ++t) s += (float)hp[t] * fb[t * 25 + l];
  bp[p * 160 + v] = (_Float16)s;
}

// ---------------- fused: dynamic filtering (K=25) + 1x1 conv GEMM (K=768, N=128)
#define BO_STRIDE 776
__global__ __launch_bounds__(256) void k_fused(
    const float* __restrict__ x, const _Float16* __restrict__ bpw,
    const _Float16* __restrict__ coef16, const float* __restrict__ bias,
    float* __restrict__ out) {
  __shared__ _Float16 xl[128 * 100];        // [c][5 rows][20 cols]
  __shared__ _Float16 bpl[16 * 152];        // [pixel][150 projected bases]
  __shared__ _Float16 bo[16 * BO_STRIDE];   // [pixel][768 = c*6+m]
  const int tid = threadIdx.x;
  const int strip = blockIdx.x;
  const int n = strip >> 8, rem = strip & 255;
  const int hh = rem >> 2, w0 = (rem & 3) << 4;
  const float* xb = x + n * (128 * 4096);
  for (int i = tid; i < 12800; i += 256) {
    int c = i / 100, r2 = i - c * 100, dr = r2 / 20, col = r2 - dr * 20;
    int yy = hh + dr - 2, xx = w0 + col - 2;
    float v = ((unsigned)yy < 64u && (unsigned)xx < 64u) ? xb[c * 4096 + yy * 64 + xx] : 0.0f;
    xl[i] = (_Float16)v;
  }
  const int pixbase = strip * 16;
  for (int i = tid; i < 16 * 150; i += 256) {
    int p = i / 150, v = i - p * 150;
    bpl[p * 152 + v] = bpw[(pixbase + p) * 160 + v];
  }
  __syncthreads();
  {
    const int p = tid & 15, cg = tid >> 4;
    _Float16 bm[150];
#pragma unroll
    for (int v = 0; v < 150; ++v) bm[v] = bpl[p * 152 + v];
    for (int i = 0; i < 8; ++i) {
      int c = cg * 8 + i;
      float pa[25];
#pragma unroll
      for (int dr = 0; dr < 5; ++dr)
#pragma unroll
        for (int dc = 0; dc < 5; ++dc)
          pa[dr * 5 + dc] = (float)xl[c * 100 + dr * 20 + (p + dc)];
#pragma unroll
      for (int m = 0; m < 6; ++m) {
        float s = 0.0f;
#pragma unroll
        for (int l = 0; l < 25; ++l) s += pa[l] * (float)bm[m * 25 + l];
        bo[p * BO_STRIDE + c * 6 + m] = (_Float16)s;
      }
    }
  }
  __syncthreads();
  const int wv = tid >> 5, lane = tid & 31;
  const int mrow = lane & 15, hi = lane >> 4;
  v8f acc = v8f{};
  const _Float16* cp = coef16 + (wv * 16 + mrow) * 768;
  for (int kc = 0; kc < 24; ++kc) {
    const int kb = kc * 32 + 8 * hi;
    __builtin_prefetch(cp + kb + 256, 0, 1);            // global_prefetch ahead on coef stream
    const _Float16* ap = &bo[mrow * BO_STRIDE + kb];
    v16h A = cat16(*(const v8h*)ap, *(const v8h*)(ap + 16));
    v16h B = cat16(*(const v8h*)(cp + kb), *(const v8h*)(cp + kb + 16));
    acc = __builtin_amdgcn_wmma_f32_16x16x32_f16(false, A, false, B,
                                                 (short)0, acc, false, false);
  }
  const int o = wv * 16 + mrow;
  const float bv = bias[o];
  float* ob = out + (n * 128 + o) * 4096 + (hh * 64 + w0);
#pragma unroll
  for (int r = 0; r < 8; ++r) {
    int prow = r + 8 * hi;
    ob[prow] = acc[r] + bv;
  }
}

extern "C" void kernel_launch(void* const* d_in, const int* in_sizes, int n_in,
                              void* d_out, int out_size, void* d_ws, size_t ws_size,
                              hipStream_t stream) {
  const float* x    = (const float*)d_in[0];
  const float* w1f  = (const float*)d_in[1];
  const float* b1   = (const float*)d_in[2];
  const float* g1   = (const float*)d_in[3];
  const float* be1  = (const float*)d_in[4];
  const float* m1   = (const float*)d_in[5];
  const float* v1   = (const float*)d_in[6];
  const float* w2f  = (const float*)d_in[7];
  const float* b2   = (const float*)d_in[8];
  const float* g2   = (const float*)d_in[9];
  const float* be2  = (const float*)d_in[10];
  const float* m2   = (const float*)d_in[11];
  const float* v2   = (const float*)d_in[12];
  const float* fb   = (const float*)d_in[13];
  const float* coef = (const float*)d_in[14];
  const float* bias = (const float*)d_in[15];
  float* out = (float*)d_out;

  _Float16* ws   = (_Float16*)d_ws;
  _Float16* w1t  = ws;                 //   73728 halves (oc, tap, c)
  _Float16* w2t  = ws + 73728;         //   41472 halves (oc, tap, c)
  _Float16* ch16 = ws + 115200;        //   98304
  _Float16* h1   = ws + 213504;        // 2097152 (32768 px * 64)
  _Float16* h2   = ws + 2310656;       // 2359296 (32768 px * 72)
  _Float16* bp   = ws + 4669952;       // 5242880 (32768 px * 160)

  k_cvt_w<<<288, 256, 0, stream>>>(w1f, w1t, 64, 128);
  k_cvt_w<<<162, 256, 0, stream>>>(w2f, w2t, 72, 64);
  k_cvt<<<384, 256, 0, stream>>>(coef, ch16, 98304);
  k_conv1<<<512, 256, 0, stream>>>(x, w1t, b1, g1, be1, m1, v1, h1);
  k_conv2<<<512, 256, 0, stream>>>(h1, w2t, b2, g2, be2, m2, v2, h2);
  k_proj<<<19200, 256, 0, stream>>>(h2, fb, bp);
  k_fused<<<2048, 256, 0, stream>>>(x, bp, ch16, bias, out);
}
